// SlotAttentionGroupingIA3Iterwise_58076547776892
// MI455X (gfx1250) — compile-verified, matrix operand order checked by reading, not run on hardware
//
#include <hip/hip_runtime.h>
#include <hip/hip_bf16.h>

#define BB 8
#define NN 4096
#define KS 24      // slots
#define FF 768
#define DD 256
#define HH 1024
#define EPSR 1e-8f

typedef __bf16 bf16;
typedef bf16  v16bf __attribute__((ext_vector_type(16)));
typedef float v8f   __attribute__((ext_vector_type(8)));
typedef unsigned int v4u __attribute__((ext_vector_type(4)));
typedef int v8i __attribute__((ext_vector_type(8)));
typedef int v4i __attribute__((ext_vector_type(4)));

__device__ __forceinline__ v8f wmma_bf16(v16bf a, v16bf b, v8f c) {
  return __builtin_amdgcn_wmma_f32_16x16x32_bf16(false, a, false, b, (short)0, c, false, false);
}

// CDNA5 16-bit A/B tile layout helpers (ISA 7.12.2):
// lane = (mn&15) | (hi<<4);  K = ((e&8)<<1) | (hi<<3) | (e&7)
__device__ __forceinline__ void ab_pos(int mn, int kk, int& lane, int& e) {
  int hi = (kk >> 3) & 1;
  lane = (mn & 15) | (hi << 4);
  e = (kk & 7) | ((kk & 16) >> 1);
}
__device__ __forceinline__ int kk_of(int lane, int e) {
  int hi = (lane >> 4) & 1;
  return ((e & 8) << 1) | (hi << 3) | (e & 7);
}

// ---------------- weight B-swizzle: B[kk=f][nn=d] = W[d*FF + f], tiles (nt over d:16, kt over f:24)
__global__ void swz_weight(const float* __restrict__ W, bf16* __restrict__ dst) {
  int idx = blockIdx.x * blockDim.x + threadIdx.x;   // 16*24*512
  const int KT = FF / 32;
  int tile = idx >> 9, r = idx & 511;
  int nt = tile / KT, kt = tile % KT;
  int lane = r >> 4, e = r & 15;
  int d = nt * 16 + (lane & 15);
  int f = kt * 32 + kk_of(lane, e);
  dst[idx] = (bf16)W[d * FF + f];
}

// ---------------- fused LayerNorm + K/V projection (bf16 WMMA), 32 rows per block.
// Emits k_sw (B-layout for dots: K=d, N=tokens) and v_sw (B-layout for updates:
// K=tokens, N=d) DIRECTLY via an LDS shuffle -- no row-major round trip to HBM.
__global__ __launch_bounds__(256) void ln_kv(
    const float* __restrict__ feat, const float* __restrict__ g, const float* __restrict__ bta,
    const bf16* __restrict__ wk_sw, const bf16* __restrict__ wv_sw,
    bf16* __restrict__ k_sw, bf16* __restrict__ v_sw) {
  __shared__ __align__(64) bf16 xs[2 * 24 * 512];   // A tiles (48 KB), reused for C shuffle
  int tid = threadIdx.x;
  int w = tid >> 5, lane = tid & 31;
  long r0 = (long)blockIdx.x * 32;
  // LN phase: each of 8 waves normalizes 4 rows (768 feats each)
  for (int rr4 = 0; rr4 < 4; ++rr4) {
    int rr = w * 4 + rr4;
    long row = r0 + rr;
    float vals[24];
    float s1 = 0.f, s2 = 0.f;
#pragma unroll
    for (int j = 0; j < 24; ++j) {
      float v = feat[row * FF + lane + 32 * j];
      vals[j] = v; s1 += v; s2 += v * v;
    }
#pragma unroll
    for (int off = 16; off >= 1; off >>= 1) {
      s1 += __shfl_xor(s1, off, 32);
      s2 += __shfl_xor(s2, off, 32);
    }
    float mean = s1 * (1.f / FF);
    float var = s2 * (1.f / FF) - mean * mean;
    float rs = rsqrtf(var + 1e-5f);
    int mt = rr >> 4, mloc = rr & 15;
#pragma unroll
    for (int j = 0; j < 24; ++j) {
      int f = lane + 32 * j;
      float xv = (vals[j] - mean) * rs * g[f] + bta[f];
      int kt = f >> 5, kkx = f & 31, ln2, e2;
      ab_pos(mloc, kkx, ln2, e2);
      xs[(mt * 24 + kt) * 512 + ln2 * 16 + e2] = (bf16)xv;
    }
  }
  __syncthreads();
  // GEMM phase: wave w owns d-tiles nt0..nt0+1, both M-tiles, both matrices
  int nt0 = w * 2;
  v8f ak[2][2] = {}, av[2][2] = {};
  for (int kt = 0; kt < 24; ++kt) {
    v16bf a0 = *(const v16bf*)&xs[(0 * 24 + kt) * 512 + lane * 16];
    v16bf a1 = *(const v16bf*)&xs[(1 * 24 + kt) * 512 + lane * 16];
#pragma unroll
    for (int j = 0; j < 2; ++j) {
      int nt = nt0 + j;
      v16bf bk = *(const v16bf*)&wk_sw[(nt * 24 + kt) * 512 + lane * 16];
      v16bf bv = *(const v16bf*)&wv_sw[(nt * 24 + kt) * 512 + lane * 16];
      ak[0][j] = wmma_bf16(a0, bk, ak[0][j]);
      ak[1][j] = wmma_bf16(a1, bk, ak[1][j]);
      av[0][j] = wmma_bf16(a0, bv, av[0][j]);
      av[1][j] = wmma_bf16(a1, bv, av[1][j]);
    }
  }
  __syncthreads();   // all waves done reading A tiles; reuse xs
  // stash C tiles (bf16) into LDS: k at xs[0..8191] as [row32][d256], v at xs[8192..]
#pragma unroll
  for (int mt = 0; mt < 2; ++mt)
#pragma unroll
    for (int j = 0; j < 2; ++j) {
      int d = (nt0 + j) * 16 + (lane & 15);
#pragma unroll
      for (int i = 0; i < 8; ++i) {
        int lr = mt * 16 + i + 8 * (lane >> 4);
        xs[lr * 256 + d] = (bf16)ak[mt][j][i];
        xs[8192 + lr * 256 + d] = (bf16)av[mt][j][i];
      }
    }
  __syncthreads();
  // emit swizzled K/V tiles (this block = 2 token N-tiles of k_sw, 1 token K-step of v_sw)
  int b = (int)(r0 / NN);
  int rloc = (int)(r0 % NN);
  int ktv = rloc >> 5;          // token k-tile index for v_sw
  int ntk = rloc >> 4;          // first token n-tile index for k_sw
  const long perK = 256L * 8 * 512;
  const long perV = 16L * 128 * 512;
  for (int ii = 0; ii < 32; ++ii) {
    int eidx = tid * 32 + ii;
    int tile = eidx >> 9, r = eidx & 511;
    int lane2 = r >> 4, e2 = r & 15;
    int kkv = kk_of(lane2, e2);
    // K: tile = mtile*8 + ktd   (B[kk=d][nn=token])
    int mtile = tile >> 3, ktd = tile & 7;
    int tokK = mtile * 16 + (lane2 & 15);
    int dK = ktd * 32 + kkv;
    k_sw[b * perK + ((long)(ntk + mtile) * 8 + ktd) * 512 + r] = xs[tokK * 256 + dK];
    // V: tile = nt over d        (B[kk=token][nn=d])
    int dV = tile * 16 + (lane2 & 15);
    v_sw[b * perV + ((long)tile * 128 + ktv) * 512 + r] = xs[8192 + kkv * 256 + dV];
  }
}

// ---------------- small dense: out[r][c] = bias[c] + slots[r]·W[c]
__global__ void slot_linear(const float* __restrict__ in, const float* __restrict__ W,
                            const float* __restrict__ bias, float* __restrict__ out, int Cout) {
  int idx = blockIdx.x * blockDim.x + threadIdx.x;
  int r = idx / Cout, c = idx % Cout;
  float s = bias[c];
  const float* ip = in + r * DD;
  const float* wp = W + (long)c * DD;
  for (int d = 0; d < DD; ++d) s += ip[d] * wp[d];
  out[idx] = s;
}

// ---------------- LN(slots) @ Wq^T * q_scale -> q in WMMA A-layout (rows>=24 zero-padded)
__global__ void ln_q(const float* __restrict__ slots, const float* __restrict__ g,
                     const float* __restrict__ bta, const float* __restrict__ Wq,
                     const float* __restrict__ qscale, bf16* __restrict__ q_sw) {
  __shared__ float sv[DD];
  __shared__ float red[256];
  int b = blockIdx.x >> 5, m = blockIdx.x & 31, t = threadIdx.x;
  float val = 0.f;
  if (m < KS) {
    int row = b * KS + m;
    float x = slots[(long)row * DD + t];
    red[t] = x; __syncthreads();
    for (int off = 128; off >= 1; off >>= 1) { if (t < off) red[t] += red[t + off]; __syncthreads(); }
    float mean = red[0] * (1.f / DD); __syncthreads();
    float dx = x - mean; red[t] = dx * dx; __syncthreads();
    for (int off = 128; off >= 1; off >>= 1) { if (t < off) red[t] += red[t + off]; __syncthreads(); }
    float var = red[0] * (1.f / DD);
    sv[t] = dx * rsqrtf(var + 1e-5f) * g[t] + bta[t];
    __syncthreads();
    float q = 0.f;
    for (int d = 0; d < DD; ++d) q += sv[d] * Wq[(long)t * DD + d];
    val = q * qscale[(long)row * DD + t];
  }
  int kt = t >> 5, kkx = t & 31, ln2, e2;
  ab_pos(m & 15, kkx, ln2, e2);
  q_sw[b * 8192 + ((m >> 4) * 8 + kt) * 512 + ln2 * 16 + e2] = (bf16)val;
}

// ---------------- dots = q·k^T * scale, softmax over slot axis (24, padded rows masked).
// Per-block: stage the whole per-batch q operand (16 KB) into LDS with one TDM
// TENSOR_LOAD_TO_LDS (1-D descriptor), then 8 waves consume it via ds loads.
__global__ __launch_bounds__(256) void dots_softmax(const bf16* __restrict__ q_sw,
                                                    const bf16* __restrict__ k_sw,
                                                    float* __restrict__ attn) {
  __shared__ __align__(64) bf16 qs[8192];                 // 16 KB staging
  int b = blockIdx.x >> 5, gblk = blockIdx.x & 31;
  int w = threadIdx.x >> 5, lane = threadIdx.x & 31;
  if (w == 0) {
    // Build Tensor DMA Descriptor (ISA 8.3/8.4): 1-D tensor, 4-byte elements,
    // tensor_dim0 = tile_dim0 = 4096 elements (16 KB), count=1, type=2.
    unsigned lds_off = (unsigned)(size_t)&qs[0];
    unsigned long long ga = (unsigned long long)(size_t)(q_sw + (long)b * 8192);
    v4u g0;
    g0.x = 1u;                                            // count=1 (user descriptor)
    g0.y = lds_off;                                       // lds_addr
    g0.z = (unsigned)(ga & 0xFFFFFFFFu);                  // global_addr[31:0]
    g0.w = (unsigned)((ga >> 32) & 0x1FFFFFFu) | (2u << 30);  // addr[56:32] | type=2
    v8i g1;
    g1[0] = (int)(2u << 16);                              // data_size=4B, no multicast
    g1[1] = (int)((4096u & 0xFFFFu) << 16);               // tensor_dim0[15:0]
    g1[2] = (int)(1u << 16);                              // tensor_dim0[31:16]=0 | tensor_dim1=1
    g1[3] = (int)(4096u << 16);                           // tile_dim0 = 4096
    g1[4] = 0;                                            // tile_dim1=0 (1-D), tile_dim2=0
    g1[5] = 4096;                                         // tensor_dim0_stride[31:0]
    g1[6] = 0;
    g1[7] = 0;
    v4i z4 = {};
    v8i z8 = {};
    __builtin_amdgcn_tensor_load_to_lds(g0, g1, z4, z4, z8, 0);
    __builtin_amdgcn_s_wait_tensorcnt(0);
  }
  __syncthreads();
  int tt = gblk * 8 + w;                                  // token tile
  const long PBk = 256L * 8 * 512;
  v8f acc0 = {}, acc1 = {};
  for (int kt = 0; kt < 8; ++kt) {
    v16bf a0 = *(const v16bf*)&qs[kt * 512 + lane * 16];
    v16bf a1 = *(const v16bf*)&qs[(8 + kt) * 512 + lane * 16];
    v16bf bm = *(const v16bf*)&k_sw[b * PBk + ((long)tt * 8 + kt) * 512 + lane * 16];
    acc0 = wmma_bf16(a0, bm, acc0);
    acc1 = wmma_bf16(a1, bm, acc1);
  }
  const float att_scale = 0.0625f;                        // 256^-0.5
  bool lowhalf = lane < 16;
  float m = -1e30f;
#pragma unroll
  for (int i = 0; i < 8; ++i) {
    acc0[i] *= att_scale; acc1[i] *= att_scale;
    m = fmaxf(m, acc0[i]);
    if (lowhalf) m = fmaxf(m, acc1[i]);                   // slots 16..23 only; 24..31 masked
  }
  m = fmaxf(m, __shfl_xor(m, 16, 32));
  float e0[8], e1[8], s = 0.f;
#pragma unroll
  for (int i = 0; i < 8; ++i) {
    e0[i] = __expf(acc0[i] - m); s += e0[i];
    e1[i] = lowhalf ? __expf(acc1[i] - m) : 0.f; s += e1[i];
  }
  s += __shfl_xor(s, 16, 32);
  float inv = 1.f / s;
  int token = tt * 16 + (lane & 15);
  int sbase = (lane >> 4) * 8;
#pragma unroll
  for (int i = 0; i < 8; ++i)
    attn[((long)(b * KS + sbase + i)) * NN + token] = e0[i] * inv;
  if (lowhalf) {
#pragma unroll
    for (int i = 0; i < 8; ++i)
      attn[((long)(b * KS + 16 + i)) * NN + token] = e1[i] * inv;
  }
}

// ---------------- per-(b,slot) row sums of attn (for eps-renorm)
__global__ void attn_rowsum(const float* __restrict__ attn, float* __restrict__ sums) {
  __shared__ float red[256];
  int bk = blockIdx.x, t = threadIdx.x;
  float s = 0.f;
  for (int n = t; n < NN; n += 256) s += attn[(long)bk * NN + n];
  red[t] = s; __syncthreads();
  for (int off = 128; off >= 1; off >>= 1) { if (t < off) red[t] += red[t + off]; __syncthreads(); }
  if (t == 0) sums[bk] = red[0] + NN * EPSR;
}

// ---------------- build renormalized attention in WMMA A-layout (K=tokens)
__global__ void build_a_sw(const float* __restrict__ attn, const float* __restrict__ sums,
                           bf16* __restrict__ a_sw) {
  long idx = (long)blockIdx.x * blockDim.x + threadIdx.x;  // B*2*128*512
  long perB = 2L * 128 * 512;
  int b = (int)(idx / perB);
  long rem = idx % perB;
  int tile = (int)(rem >> 9), r = (int)(rem & 511);
  int mt = tile / 128, kt = tile % 128;
  int lane = r >> 4, e = r & 15;
  int m = mt * 16 + (lane & 15);
  int token = kt * 32 + kk_of(lane, e);
  float val = 0.f;
  if (m < KS) {
    int row = b * KS + m;
    val = (attn[(long)row * NN + token] + EPSR) / sums[row];
  }
  a_sw[idx] = (bf16)val;
}

// ---------------- updates = a @ v (bf16 WMMA), scaled by v_scale
__global__ __launch_bounds__(256) void updates_gemm(const bf16* __restrict__ a_sw,
                                                    const bf16* __restrict__ v_sw,
                                                    const float* __restrict__ vscale,
                                                    float* __restrict__ upd) {
  int b = blockIdx.x;
  int w = threadIdx.x >> 5, lane = threadIdx.x & 31;
  int nt0 = w * 2;
  const long perA = 2L * 128 * 512;
  const long perV = 16L * 128 * 512;
  v8f u[2][2] = {};
  for (int kt = 0; kt < 128; ++kt) {
    v16bf a0 = *(const v16bf*)&a_sw[b * perA + (long)kt * 512 + lane * 16];
    v16bf a1 = *(const v16bf*)&a_sw[b * perA + (long)(128 + kt) * 512 + lane * 16];
#pragma unroll
    for (int j = 0; j < 2; ++j) {
      v16bf bm = *(const v16bf*)&v_sw[b * perV + ((long)(nt0 + j) * 128 + kt) * 512 + lane * 16];
      u[0][j] = wmma_bf16(a0, bm, u[0][j]);
      u[1][j] = wmma_bf16(a1, bm, u[1][j]);
    }
  }
#pragma unroll
  for (int mt = 0; mt < 2; ++mt)
#pragma unroll
    for (int j = 0; j < 2; ++j) {
      int d = (nt0 + j) * 16 + (lane & 15);
#pragma unroll
      for (int i = 0; i < 8; ++i) {
        int slot = mt * 16 + i + 8 * (lane >> 4);
        if (slot < KS) {
          long o = ((long)(b * KS + slot)) * DD + d;
          upd[o] = u[mt][j][i] * vscale[o];
        }
      }
    }
}

// ---------------- GRU cell + FF (IA3-scaled hidden) + residual, one (b,slot) row per block
__global__ void gru_ff(const float* __restrict__ upd, const float* __restrict__ hprev_g,
                       const float* __restrict__ wih, const float* __restrict__ whh,
                       const float* __restrict__ bih, const float* __restrict__ bhh,
                       const float* __restrict__ lng, const float* __restrict__ lnb,
                       const float* __restrict__ W1, const float* __restrict__ b1,
                       const float* __restrict__ W2, const float* __restrict__ b2,
                       const float* __restrict__ ffscale,
                       float* __restrict__ slots_next, float* __restrict__ out2) {
  __shared__ float su[DD], sh[DD], lnv[DD], red[256];
  __shared__ float hb[HH];
  int row = blockIdx.x, t = threadIdx.x;
  su[t] = upd[(long)row * DD + t];
  sh[t] = hprev_g[(long)row * DD + t];
  __syncthreads();
  float xr = bih[t], xz = bih[DD + t], xn = bih[2 * DD + t];
  float hr = bhh[t], hz = bhh[DD + t], hn = bhh[2 * DD + t];
  for (int d = 0; d < DD; ++d) {
    float ud = su[d], hd = sh[d];
    xr += wih[(long)t * DD + d] * ud;
    xz += wih[((long)(DD + t)) * DD + d] * ud;
    xn += wih[((long)(2 * DD + t)) * DD + d] * ud;
    hr += whh[(long)t * DD + d] * hd;
    hz += whh[((long)(DD + t)) * DD + d] * hd;
    hn += whh[((long)(2 * DD + t)) * DD + d] * hd;
  }
  float r = 1.f / (1.f + __expf(-(xr + hr)));
  float z = 1.f / (1.f + __expf(-(xz + hz)));
  float n_ = tanhf(xn + r * hn);
  float sn = (1.f - z) * n_ + z * sh[t];
  // LayerNorm over new slots
  red[t] = sn; __syncthreads();
  for (int off = 128; off >= 1; off >>= 1) { if (t < off) red[t] += red[t + off]; __syncthreads(); }
  float mean = red[0] * (1.f / DD); __syncthreads();
  float dx = sn - mean; red[t] = dx * dx; __syncthreads();
  for (int off = 128; off >= 1; off >>= 1) { if (t < off) red[t] += red[t + off]; __syncthreads(); }
  float var = red[0] * (1.f / DD);
  lnv[t] = dx * rsqrtf(var + 1e-5f) * lng[t] + lnb[t];
  __syncthreads();
#pragma unroll
  for (int j = 0; j < 4; ++j) {
    int c = j * 256 + t;
    float hv = b1[c];
    for (int d = 0; d < DD; ++d) hv += lnv[d] * W1[(long)c * DD + d];
    hb[c] = fmaxf(hv, 0.f) * ffscale[(long)row * HH + c];
  }
  __syncthreads();
  float o = b2[t];
  for (int c = 0; c < HH; ++c) o += hb[c] * W2[(long)t * HH + c];
  float res = sn + o;
  slots_next[(long)row * DD + t] = res;
  if (out2) out2[(long)row * DD + t] = res;
}

extern "C" void kernel_launch(void* const* d_in, const int* in_sizes, int n_in,
                              void* d_out, int out_size, void* d_ws, size_t ws_size,
                              hipStream_t stream) {
  const float* features  = (const float*)d_in[0];
  const float* cond      = (const float*)d_in[1];
  const float* norm_in_g = (const float*)d_in[2];
  const float* norm_in_b = (const float*)d_in[3];
  const float* Wk        = (const float*)d_in[4];
  const float* Wv        = (const float*)d_in[5];
  const float* norm_s_g  = (const float*)d_in[6];
  const float* norm_s_b  = (const float*)d_in[7];
  const float* Wq        = (const float*)d_in[8];
  const float* q_sW      = (const float*)d_in[9];
  const float* q_sb      = (const float*)d_in[10];
  const float* v_sW      = (const float*)d_in[11];
  const float* v_sb      = (const float*)d_in[12];
  const float* f_sW      = (const float*)d_in[13];
  const float* f_sb      = (const float*)d_in[14];
  const float* gru_wih   = (const float*)d_in[15];
  const float* gru_whh   = (const float*)d_in[16];
  const float* gru_bih   = (const float*)d_in[17];
  const float* gru_bhh   = (const float*)d_in[18];
  const float* ffln_g    = (const float*)d_in[19];
  const float* ffln_b    = (const float*)d_in[20];
  const float* W1        = (const float*)d_in[21];
  const float* b1        = (const float*)d_in[22];
  const float* W2        = (const float*)d_in[23];
  const float* b2        = (const float*)d_in[24];

  char* ws = (char*)d_ws;
  size_t off = 0;
  auto alloc = [&](size_t bytes) { size_t o = off; off = (off + bytes + 255) & ~(size_t)255; return o; };
  bf16*  wk_sw   = (bf16*)(ws + alloc(16L * 24 * 512 * 2));
  bf16*  wv_sw   = (bf16*)(ws + alloc(16L * 24 * 512 * 2));
  bf16*  k_sw    = (bf16*)(ws + alloc(8L * 256 * 8 * 512 * 2));
  bf16*  v_sw    = (bf16*)(ws + alloc(8L * 16 * 128 * 512 * 2));
  bf16*  q_sw    = (bf16*)(ws + alloc((long)BB * 8192 * 2));
  bf16*  a_sw    = (bf16*)(ws + alloc(8L * 2 * 128 * 512 * 2));
  float* qscale  = (float*)(ws + alloc((long)BB * KS * DD * 4));
  float* vscale  = (float*)(ws + alloc((long)BB * KS * DD * 4));
  float* ffscale = (float*)(ws + alloc((long)BB * KS * HH * 4));
  float* sums    = (float*)(ws + alloc((long)BB * KS * 4));
  float* upd     = (float*)(ws + alloc((long)BB * KS * DD * 4));
  float* slots0  = (float*)(ws + alloc((long)BB * KS * DD * 4));
  float* slots1  = (float*)(ws + alloc((long)BB * KS * DD * 4));

  float* outslots = (float*)d_out;
  float* attn     = (float*)d_out + (size_t)BB * KS * DD;

  // slots = conditioning
  (void)hipMemcpyAsync(slots0, cond, (size_t)BB * KS * DD * 4, hipMemcpyDeviceToDevice, stream);

  // iteration-invariant: swizzled weights, fused LN + K/V projection emitting
  // WMMA-ready operand layouts directly (no row-major k/v round trip)
  swz_weight<<<768, 256, 0, stream>>>(Wk, wk_sw);
  swz_weight<<<768, 256, 0, stream>>>(Wv, wv_sw);
  ln_kv<<<BB * NN / 32, 256, 0, stream>>>(features, norm_in_g, norm_in_b, wk_sw, wv_sw, k_sw, v_sw);

  float* slots_cur = slots0;
  float* slots_nxt = slots1;
  for (int it = 0; it < 3; ++it) {
    slot_linear<<<192, 256, 0, stream>>>(slots_cur, q_sW, q_sb, qscale, DD);
    slot_linear<<<192, 256, 0, stream>>>(slots_cur, v_sW, v_sb, vscale, DD);
    slot_linear<<<768, 256, 0, stream>>>(slots_cur, f_sW, f_sb, ffscale, HH);
    ln_q<<<BB * 32, 256, 0, stream>>>(slots_cur, norm_s_g, norm_s_b, Wq, qscale, q_sw);
    dots_softmax<<<BB * 32, 256, 0, stream>>>(q_sw, k_sw, attn);
    attn_rowsum<<<BB * KS, 256, 0, stream>>>(attn, sums);
    build_a_sw<<<4096, 256, 0, stream>>>(attn, sums, a_sw);
    updates_gemm<<<BB, 256, 0, stream>>>(a_sw, v_sw, vscale, upd);
    gru_ff<<<BB * KS, 256, 0, stream>>>(upd, slots_cur, gru_wih, gru_whh, gru_bih, gru_bhh,
                                        ffln_g, ffln_b, W1, b1, W2, b2, ffscale,
                                        slots_nxt, (it == 2) ? outslots : nullptr);
    float* tmp = slots_cur; slots_cur = slots_nxt; slots_nxt = tmp;
  }
}
